// AblationAttention_62801011802507
// MI455X (gfx1250) — matrix operand order repaired and checked by reading, not compiled, not run on hardware
//
#include <hip/hip_runtime.h>

#define DEV static __device__ __forceinline__

typedef __attribute__((ext_vector_type(16))) __bf16 v16bf;
typedef __attribute__((ext_vector_type(8)))  __bf16 v8bf;
typedef __attribute__((ext_vector_type(8)))  float  v8f;
typedef __attribute__((ext_vector_type(4)))  float  f32x4;
typedef __attribute__((ext_vector_type(4)))  unsigned int u32x4;
typedef __attribute__((ext_vector_type(8)))  int    i32x8;
typedef __attribute__((ext_vector_type(4)))  int    i32x4;

namespace {
constexpr int kB = 2, kS = 2048, kE = 1024, kH = 16, kD = 64;
constexpr int kBH = kB * kH;   // 32
constexpr int kBS = kB * kS;   // 4096
}

// ---------- bf16 helpers: use native HW converters ----------
DEV __bf16 f2bf(float f) { return (__bf16)f; }

// ---------- WMMA wrapper ----------
DEV v8f wmma_bf16(v16bf a, v16bf b, v8f c) {
    return __builtin_amdgcn_wmma_f32_16x16x32_bf16(
        /*neg_a=*/false, a, /*neg_b=*/false, b,
        /*c_mod=*/(short)0, c, /*reuse_a=*/false, /*reuse_b=*/false);
}

// ---------- fragment loaders (this code's contiguous k = 16*hf + s for A and B, wave32) ----------
DEV v16bf load_a16(const __bf16* p0, const __bf16* p1) {
    v8bf x = *(const v8bf*)p0, y = *(const v8bf*)p1;
    v16bf a;
#pragma unroll
    for (int t = 0; t < 8; ++t) { a[t] = x[t]; a[8 + t] = y[t]; }
    return a;
}
DEV v16bf load_b16(const __bf16* p) {
    v8bf x = *(const v8bf*)p, y = *(const v8bf*)(p + 8);
    v16bf b;
#pragma unroll
    for (int t = 0; t < 8; ++t) { b[t] = x[t]; b[8 + t] = y[t]; }
    return b;
}

// ---------- TDM descriptor helper: 2D tile load Global -> LDS ----------
// data_size = 2 bytes. tile_d0 elements contiguous per row, tile_d1 rows,
// row stride = stride_el (elements). LDS rows padded: after pad_int_code
// DWORDs (0=2,1=4,2=8,3=16,4=32,...) insert (pad_amt_code+1) DWORDs.
DEV void tdm_load_2d(unsigned lds_off, const void* gaddr,
                     int tile_d0, int tile_d1, int stride_el, int tdim1,
                     int pad_int_code, int pad_amt_code) {
    unsigned long long ga = (unsigned long long)(uintptr_t)gaddr;
    u32x4 g0;
    g0[0] = 1u;                                   // count=1, is_restore=0
    g0[1] = lds_off;                              // lds_addr (bytes)
    g0[2] = (unsigned)(ga & 0xffffffffu);         // global_addr[31:0]
    g0[3] = (unsigned)((ga >> 32) & 0x1ffffffu)   // global_addr[56:32]
          | (2u << 30);                           // type=2 (image)
    i32x8 g1;
    g1[0] = (1 << 16)                             // data_size = 2B
          | (1 << 20)                             // pad_enable
          | (pad_int_code << 22) | (pad_amt_code << 25);
    g1[1] = (stride_el & 0xffff) << 16;           // tensor_dim0 low16 @ [63:48]
    g1[2] = ((stride_el >> 16) & 0xffff)          // tensor_dim0 high16
          | ((tdim1 & 0xffff) << 16);             // tensor_dim1 low16
    g1[3] = ((tdim1 >> 16) & 0xffff)              // tensor_dim1 high16
          | ((tile_d0 & 0xffff) << 16);           // tile_dim0
    g1[4] = (tile_d1 & 0xffff);                   // tile_dim1 (tile_dim2=0)
    g1[5] = stride_el;                            // tensor_dim0_stride low32
    g1[6] = 0;                                    // stride hi, dim1_stride lo
    g1[7] = 0;                                    // dim1_stride hi
    i32x4 z4 = {0, 0, 0, 0};
    i32x8 z8 = {0, 0, 0, 0, 0, 0, 0, 0};
    __builtin_amdgcn_tensor_load_to_lds(g0, g1, z4, z4, z8, 0);
}

// =====================================================================
// Kernel 0: fp32 -> bf16 bulk convert (n multiple of 2048)
// =====================================================================
__global__ __launch_bounds__(256) void k_cvt(const float* __restrict__ src,
                                             __bf16* __restrict__ dst, int n) {
    const int idx = (blockIdx.x * 256 + threadIdx.x) * 8;
    if (idx >= n) return;
    f32x4 a = *(const f32x4*)(src + idx);
    f32x4 b = *(const f32x4*)(src + idx + 4);
    v8bf o;
#pragma unroll
    for (int t = 0; t < 4; ++t) { o[t] = f2bf(a[t]); o[4 + t] = f2bf(b[t]); }
    *(v8bf*)(dst + idx) = o;
}

// =====================================================================
// Kernel 1: v = x @ wv_w^T + wv_b   -> Vf laid out [b][h][s][d] (fp32)
// Wave tile 32(M) x 64(N): 2 A-frags x 4 B-frags = 8 WMMAs / k-step.
// =====================================================================
__global__ __launch_bounds__(128) void k_gemm_v(const __bf16* __restrict__ X,
                                                const __bf16* __restrict__ W,
                                                const float* __restrict__ bias,
                                                float* __restrict__ Vf) {
    const int lane = threadIdx.x & 31, wave = threadIdx.x >> 5;
    const int hf = lane >> 4, ln = lane & 15;
    const int gw = blockIdx.x * 4 + wave;
    const int tilesN = kE / 64;                 // 16
    const int mt = gw / tilesN, nt = gw % tilesN;
    const int m0 = mt * 32, n0 = nt * 64;

    v8f acc0[4] = {}, acc1[4] = {};
    const __bf16* arow0 = X + (size_t)(m0 + ln) * kE;
    const __bf16* arow1 = X + (size_t)(m0 + 16 + ln) * kE;
    for (int k0 = 0; k0 < kE; k0 += 32) {
        v16bf a0 = load_a16(arow0 + k0 + hf * 16, arow0 + k0 + hf * 16 + 8);
        v16bf a1 = load_a16(arow1 + k0 + hf * 16, arow1 + k0 + hf * 16 + 8);
#pragma unroll
        for (int t = 0; t < 4; ++t) {
            const __bf16* wrow = W + (size_t)(n0 + t * 16 + ln) * kE + k0 + hf * 16;
            v16bf b = load_b16(wrow);
            acc0[t] = wmma_bf16(a0, b, acc0[t]);
            acc1[t] = wmma_bf16(a1, b, acc1[t]);
        }
    }
#pragma unroll
    for (int t = 0; t < 4; ++t) {
        const int col = n0 + t * 16 + ln;
        const float bv = bias[col];
        const int hh = col >> 6, dd = col & 63;
#pragma unroll
        for (int r = 0; r < 8; ++r) {
            {
                const int row = m0 + hf * 8 + r;
                const int bb = row >> 11, ss = row & (kS - 1);
                Vf[(((size_t)bb * kH + hh) * kS + ss) * kD + dd] = acc0[t][r] + bv;
            }
            {
                const int row = m0 + 16 + hf * 8 + r;
                const int bb = row >> 11, ss = row & (kS - 1);
                Vf[(((size_t)bb * kH + hh) * kS + ss) * kD + dd] = acc1[t][r] + bv;
            }
        }
    }
}

// =====================================================================
// Kernel 2a/2b: per-batch min/max (two-stage deterministic reduction)
// =====================================================================
__global__ __launch_bounds__(256) void k_minmax1(const float* __restrict__ Vf,
                                                 float* __restrict__ pmin,
                                                 float* __restrict__ pmax) {
    __shared__ float smin[256], smax[256];
    const int tid = threadIdx.x;
    const int batch = blockIdx.x >> 8, chunk = blockIdx.x & 255;
    const size_t base = (size_t)batch * (kH * kS * kD) + (size_t)chunk * 8192;
    float mn = 3.0e38f, mx = -3.0e38f;
    for (int t = tid; t < 8192; t += 256) {
        float v = Vf[base + t];
        mn = fminf(mn, v); mx = fmaxf(mx, v);
    }
    smin[tid] = mn; smax[tid] = mx; __syncthreads();
    for (int s = 128; s > 0; s >>= 1) {
        if (tid < s) {
            smin[tid] = fminf(smin[tid], smin[tid + s]);
            smax[tid] = fmaxf(smax[tid], smax[tid + s]);
        }
        __syncthreads();
    }
    if (tid == 0) { pmin[blockIdx.x] = smin[0]; pmax[blockIdx.x] = smax[0]; }
}

__global__ __launch_bounds__(256) void k_minmax2(const float* __restrict__ pmin,
                                                 const float* __restrict__ pmax,
                                                 float* __restrict__ vmm) {
    __shared__ float smin[256], smax[256];
    const int tid = threadIdx.x, b = blockIdx.x;
    smin[tid] = pmin[b * 256 + tid];
    smax[tid] = pmax[b * 256 + tid];
    __syncthreads();
    for (int s = 128; s > 0; s >>= 1) {
        if (tid < s) {
            smin[tid] = fminf(smin[tid], smin[tid + s]);
            smax[tid] = fmaxf(smax[tid], smax[tid + s]);
        }
        __syncthreads();
    }
    if (tid == 0) { vmm[b * 2] = smin[0]; vmm[b * 2 + 1] = smax[0]; }
}

// =====================================================================
// Kernel 3: per (b,h) causal running-mean scan.
// =====================================================================
__global__ __launch_bounds__(32) void k_scanqk(const float* __restrict__ Vf,
                                               const float* __restrict__ vmm,
                                               __bf16* __restrict__ Q,
                                               __bf16* __restrict__ K) {
    const int bh = blockIdx.x;
    const int b = bh >> 4;
    const int lane = threadIdx.x;
    const float vmin = vmm[b * 2];
    const float vmax = vmm[b * 2 + 1];
    const float invr = 1.0f / (vmax - vmin + 1e-8f);
    const size_t base = (size_t)bh * kS * kD;
    float acc0 = 0.f, acc1 = 0.f;
    for (int i = 0; i < kS; ++i) {
        const size_t off = base + (size_t)i * kD;
        const float v0 = Vf[off + lane], v1 = Vf[off + lane + 32];
        const float vs0 = (v0 - vmin) * invr, vs1 = (v1 - vmin) * invr;
        K[off + lane] = f2bf(vs0);
        K[off + lane + 32] = f2bf(vs1);
        acc0 += vs0; acc1 += vs1;
        const float cinv = 1.0f / (float)(i + 1);
        const float d0 = acc0 * cinv + 0.5f, d1 = acc1 * cinv + 0.5f;
        const float r0 = 1.0f / d0, r1 = 1.0f / d1;
        float si = r0 + r1;
        si += __shfl_xor(si, 1, 32);
        si += __shfl_xor(si, 2, 32);
        si += __shfl_xor(si, 4, 32);
        si += __shfl_xor(si, 8, 32);
        si += __shfl_xor(si, 16, 32);
        const float isi = 1.0f / si;
        Q[off + lane] = f2bf(r0 * isi);
        Q[off + lane + 32] = f2bf(r1 * isi);
    }
}

// =====================================================================
// Kernel 3b: V transpose+convert: Vf[bh][s][d] (f32) -> Vt[bh][d][s] (bf16)
// =====================================================================
__global__ __launch_bounds__(256) void k_transpose(const float* __restrict__ Vf,
                                                   __bf16* __restrict__ Vt) {
    constexpr int TST = 72;
    __shared__ __align__(16) __bf16 T[64 * TST];
    const int bh = blockIdx.y;
    const int s0 = blockIdx.x * 64;
    const int tid = threadIdx.x;
    {
        const int s = tid >> 2, dblk = (tid & 3) << 4;
        const float* src = Vf + ((size_t)bh * kS + s0 + s) * kD + dblk;
#pragma unroll
        for (int q = 0; q < 16; q += 4) {
            f32x4 v = *(const f32x4*)(src + q);
#pragma unroll
            for (int p = 0; p < 4; ++p) T[s * TST + dblk + q + p] = f2bf(v[p]);
        }
    }
    __syncthreads();
    {
        const int d = tid >> 2, jblk = (tid & 3) << 4;
        v8bf o0, o1;
#pragma unroll
        for (int q = 0; q < 8; ++q) {
            o0[q] = T[(jblk + q) * TST + d];
            o1[q] = T[(jblk + 8 + q) * TST + d];
        }
        __bf16* dst = Vt + ((size_t)bh * kD + d) * kS + s0 + jblk;
        *(v8bf*)dst = o0;
        *(v8bf*)(dst + 8) = o1;
    }
}

// =====================================================================
// Kernel 4: causal flash attention with WMMA + TDM staging.
//  grid = (S/64, B*H), block = 128 (4 waves). Wave 0 issues two
//  tensor_load_to_lds per 32-key chunk (K tile 32x64, Vt tile 64x32,
//  LDS-padded strides), waits on TENSORcnt, block barrier publishes.
// =====================================================================
__global__ __launch_bounds__(128) void k_flash(const __bf16* __restrict__ Q,
                                               const __bf16* __restrict__ K,
                                               const __bf16* __restrict__ Vt,
                                               __bf16* __restrict__ attn) {
    constexpr int KST = 72;   // halves per K row  = 64 + 8 pad (4 DW per 32 DW)
    constexpr int VST = 40;   // halves per Vt row = 32 + 8 pad (4 DW per 16 DW)
    constexpr int PST = 40;   // halves per P row
    __shared__ __align__(16) __bf16 Ksh[32 * KST];
    __shared__ __align__(16) __bf16 Vtsh[64 * VST];
    __shared__ __align__(16) __bf16 Psh[4 * 16 * PST];

    const int bh = blockIdx.y;
    const int b = bh >> 4, h = bh & 15;
    const int tid = threadIdx.x;
    const int lane = tid & 31, wave = tid >> 5;
    const int hf = lane >> 4, ln = lane & 15;
    const int i0 = blockIdx.x * 64 + wave * 16;
    __bf16* Pw = Psh + wave * 16 * PST;
    const unsigned koff = (unsigned)(uintptr_t)&Ksh[0];   // LDS byte offsets
    const unsigned voff = (unsigned)(uintptr_t)&Vtsh[0];

    // Q fragments (16 rows x 64 d) -> two A-frags
    const __bf16* qrow = Q + ((size_t)bh * kS + i0 + ln) * kD;
    const v16bf qa0 = load_a16(qrow + hf * 16,      qrow + hf * 16 + 8);
    const v16bf qa1 = load_a16(qrow + 32 + hf * 16, qrow + 40 + hf * 16);

    v8f o[4] = {};
    float rmax[8], rsum[8];
#pragma unroll
    for (int r = 0; r < 8; ++r) { rmax[r] = -3.0e38f; rsum[r] = 0.f; }

    const int jcmax = blockIdx.x * 64 + 63;
    for (int jc = 0; jc <= jcmax; jc += 32) {
        __syncthreads();   // previous chunk's compute done before restage
        if (wave == 0) {
            // K tile: 32 rows (j) x 64 contiguous halves (d); row stride 64
            tdm_load_2d(koff, K + ((size_t)bh * kS + jc) * kD,
                        /*tile_d0=*/64, /*tile_d1=*/32, /*stride=*/kD,
                        /*tdim1=*/kS, /*pad_int(32DW)=*/4, /*pad_amt(4DW)=*/3);
            // Vt tile: 64 rows (d) x 32 contiguous halves (j); row stride kS
            tdm_load_2d(voff, Vt + ((size_t)bh * kD) * kS + jc,
                        /*tile_d0=*/32, /*tile_d1=*/64, /*stride=*/kS,
                        /*tdim1=*/kD, /*pad_int(16DW)=*/3, /*pad_amt(4DW)=*/3);
            __builtin_amdgcn_s_wait_tensorcnt(0);
        }
        __syncthreads();   // staged tiles visible to all waves

        if (jc > i0 + 15) continue;   // wave-uniform: fully masked chunk

        // ---- scores: two 16x16 j-tiles, K-dim = D = 64 (2 WMMAs each)
        v8f s0 = {}, s1 = {};
        {
            v16bf b00 = load_b16(Ksh + ln * KST + hf * 16);            // j0, d 0..31
            v16bf b01 = load_b16(Ksh + ln * KST + 32 + hf * 16);       // j0, d 32..63
            s0 = wmma_bf16(qa0, b00, s0);
            s0 = wmma_bf16(qa1, b01, s0);
            v16bf b10 = load_b16(Ksh + (16 + ln) * KST + hf * 16);     // j1, d 0..31
            v16bf b11 = load_b16(Ksh + (16 + ln) * KST + 32 + hf * 16);// j1, d 32..63
            s1 = wmma_bf16(qa0, b10, s1);
            s1 = wmma_bf16(qa1, b11, s1);
        }

        // ---- causal mask + online softmax (row = r + 8*hf, col = ln)
#pragma unroll
        for (int r = 0; r < 8; ++r) {
            const int irow = i0 + hf * 8 + r;
            float a0 = s0[r]; if (jc + ln > irow)      a0 = -3.0e38f;
            float a1 = s1[r]; if (jc + 16 + ln > irow) a1 = -3.0e38f;
            float tm = fmaxf(a0, a1);
            tm = fmaxf(tm, __shfl_xor(tm, 1, 32));
            tm = fmaxf(tm, __shfl_xor(tm, 2, 32));
            tm = fmaxf(tm, __shfl_xor(tm, 4, 32));
            tm = fmaxf(tm, __shfl_xor(tm, 8, 32));
            const float nm = fmaxf(rmax[r], tm);
            const float sc = __expf(rmax[r] - nm);
            const float p0 = __expf(a0 - nm);
            const float p1 = __expf(a1 - nm);
            float ps = p0 + p1;
            ps += __shfl_xor(ps, 1, 32);
            ps += __shfl_xor(ps, 2, 32);
            ps += __shfl_xor(ps, 4, 32);
            ps += __shfl_xor(ps, 8, 32);
            rsum[r] = rsum[r] * sc + ps;
            rmax[r] = nm;
            o[0][r] *= sc; o[1][r] *= sc; o[2][r] *= sc; o[3][r] *= sc;
            Pw[(hf * 8 + r) * PST + ln]      = f2bf(p0);
            Pw[(hf * 8 + r) * PST + ln + 16] = f2bf(p1);
        }

        // ---- re-read P in A-fragment layout (wave-private LDS pad)
        const __bf16* pr = Pw + ln * PST;
        v16bf pa = load_a16(pr + hf * 16, pr + hf * 16 + 8);

        // ---- PV: 4 d-tiles, K-dim = 32 keys
#pragma unroll
        for (int dt = 0; dt < 4; ++dt) {
            v16bf vb = load_b16(Vtsh + (dt * 16 + ln) * VST + hf * 16);
            o[dt] = wmma_bf16(pa, vb, o[dt]);
        }
    }

    // ---- normalize and write bf16 attention output (row-major B*S x E)
#pragma unroll
    for (int r = 0; r < 8; ++r) {
        const float inv = 1.0f / rsum[r];
        const int irow = i0 + hf * 8 + r;
        __bf16* dst = attn + ((size_t)(b * kS + irow)) * kE + h * kD;
#pragma unroll
        for (int dt = 0; dt < 4; ++dt)
            dst[dt * 16 + ln] = f2bf(o[dt][r] * inv);
    }
}

// =====================================================================
// Kernel 5: out = attn(bf16) @ wo_w^T + wo_b  (fp32 output)
// =====================================================================
__global__ __launch_bounds__(128) void k_gemm_o(const __bf16* __restrict__ A,
                                                const __bf16* __restrict__ W,
                                                const float* __restrict__ bias,
                                                float* __restrict__ out) {
    const int lane = threadIdx.x & 31, wave = threadIdx.x >> 5;
    const int hf = lane >> 4, ln = lane & 15;
    const int gw = blockIdx.x * 4 + wave;
    const int tilesN = kE / 64;
    const int mt = gw / tilesN, nt = gw % tilesN;
    const int m0 = mt * 32, n0 = nt * 64;

    v8f acc0[4] = {}, acc1[4] = {};
    const __bf16* arow0 = A + (size_t)(m0 + ln) * kE;
    const __bf16* arow1 = A + (size_t)(m0 + 16 + ln) * kE;
    for (int k0 = 0; k0 < kE; k0 += 32) {
        v16bf a0 = load_a16(arow0 + k0 + hf * 16, arow0 + k0 + hf * 16 + 8);
        v16bf a1 = load_a16(arow1 + k0 + hf * 16, arow1 + k0 + hf * 16 + 8);
#pragma unroll
        for (int t = 0; t < 4; ++t) {
            const __bf16* wrow = W + (size_t)(n0 + t * 16 + ln) * kE + k0 + hf * 16;
            v16bf b = load_b16(wrow);
            acc0[t] = wmma_bf16(a0, b, acc0[t]);
            acc1[t] = wmma_bf16(a1, b, acc1[t]);
        }
    }
#pragma unroll
    for (int t = 0; t < 4; ++t) {
        const int col = n0 + t * 16 + ln;
        const float bv = bias[col];
#pragma unroll
        for (int r = 0; r < 8; ++r) {
            out[(size_t)(m0 + hf * 8 + r) * kE + col] = acc0[t][r] + bv;
            out[(size_t)(m0 + 16 + hf * 8 + r) * kE + col] = acc1[t][r] + bv;
        }
    }
}

// =====================================================================
extern "C" void kernel_launch(void* const* d_in, const int* in_sizes, int n_in,
                              void* d_out, int out_size, void* d_ws, size_t ws_size,
                              hipStream_t stream) {
    (void)in_sizes; (void)n_in; (void)out_size; (void)ws_size;
    const float* x    = (const float*)d_in[0];
    const float* wv_w = (const float*)d_in[1];
    const float* wv_b = (const float*)d_in[2];
    const float* wo_w = (const float*)d_in[3];
    const float* wo_b = (const float*)d_in[4];
    float* out = (float*)d_out;

    char* ws = (char*)d_ws;
    size_t off = 0;
    float*  Vf  = (float*)(ws + off);  off += (size_t)kBH * kS * kD * sizeof(float); // 16 MB
    __bf16* Qb  = (__bf16*)(ws + off); off += (size_t)kBH * kS * kD * 2;             // 8 MB
    __bf16* Kb  = (__bf16*)(ws + off); off += (size_t)kBH * kS * kD * 2;             // 8 MB
    __bf16* VtG = (__bf16*)(ws + off); off += (size_t)kBH * kS * kD * 2;             // 8 MB
    __bf16* Xb  = (__bf16*)(ws + off);            // alias: dead after k_gemm_v
    __bf16* Ab  = (__bf16*)(ws + off); off += (size_t)kBS * kE * 2;                  // 8 MB
    __bf16* Wvb = (__bf16*)(ws + off); off += (size_t)kE * kE * 2;                   // 2 MB
    __bf16* Wob = (__bf16*)(ws + off); off += (size_t)kE * kE * 2;                   // 2 MB
    float* pmin = (float*)(ws + off);  off += 512 * sizeof(float);
    float* pmax = (float*)(ws + off);  off += 512 * sizeof(float);
    float* vmm  = (float*)(ws + off);  off += 4 * sizeof(float);

    const int nX = kBS * kE, nW = kE * kE;
    k_cvt<<<nX / 2048, 256, 0, stream>>>(x, Xb, nX);
    k_cvt<<<nW / 2048, 256, 0, stream>>>(wv_w, Wvb, nW);
    k_cvt<<<nW / 2048, 256, 0, stream>>>(wo_w, Wob, nW);
    k_gemm_v<<<512, 128, 0, stream>>>(Xb, Wvb, wv_b, Vf);
    k_minmax1<<<512, 256, 0, stream>>>(Vf, pmin, pmax);
    k_minmax2<<<kB, 256, 0, stream>>>(pmin, pmax, vmm);
    k_scanqk<<<kBH, 32, 0, stream>>>(Vf, vmm, Qb, Kb);
    k_transpose<<<dim3(kS / 64, kBH), 256, 0, stream>>>(Vf, VtG);
    k_flash<<<dim3(kS / 64, kBH), 128, 0, stream>>>(Qb, Kb, VtG, Ab);
    k_gemm_o<<<512, 128, 0, stream>>>(Ab, Wob, wo_b, out);
}
